// SnailDenseBlock_35433480192929
// MI455X (gfx1250) — compile-verified, matrix operand order checked by reading, not run on hardware
//
#include <hip/hip_runtime.h>

// ---------------------------------------------------------------------------
// SnailDenseBlock on gfx1250 (MI455X):
//   out[:, :, 0:256]   = tanh(conv_v(x)) * sigmoid(conv_g(x))
//   out[:, :, 256:512] = x
// One M=131072, K=512, N=512 GEMM (N = [value|gate]) on
// v_wmma_f32_16x16x32_bf16, f32 accumulate, fused bias+tanh*sigmoid epilogue.
// Roofline: ~69 GFLOP vs ~400 MB HBM -> memory floor ~17us @ 23.3 TB/s; only
// the bf16 WMMA path (16384 FLOP/instr) keeps this memory bound. Output is
// written with TH=NT store hints so the one-shot 268 MB out-stream does not
// evict the reused x-stream (134 MB, fits the 192 MB L2).
// ---------------------------------------------------------------------------

typedef __attribute__((ext_vector_type(16))) __bf16 v16bf;
typedef __attribute__((ext_vector_type(8)))  float  v8f;
typedef __attribute__((ext_vector_type(4)))  float  v4f;

#define B_   32
#define T_   4096
#define C_   256
#define F_   256
#define K_   512   // 2*C  (prev-tap | cur-tap channels)
#define N_   512   // 2*F  (value | gate columns)
#define OUTW 512

__device__ __forceinline__ unsigned short f2bf(float f) {
    union { float f; unsigned u; } v; v.f = f;
    unsigned u = v.u;
    unsigned r = (u + 0x7FFFu + ((u >> 16) & 1u)) >> 16;  // RNE
    return (unsigned short)r;
}

// HW transcendental tanh on CDNA5 (TRANS op class); guarded fallback via exp2.
__device__ __forceinline__ float fast_tanh(float x) {
#if __has_builtin(__builtin_amdgcn_tanhf)
    return __builtin_amdgcn_tanhf(x);
#else
    float e = __builtin_amdgcn_exp2f(x * 2.8853900817779268f);   // exp(2x)
    return 1.0f - 2.0f * __builtin_amdgcn_rcpf(e + 1.0f);
#endif
}

// ---------------------------------------------------------------------------
// Pack combined B matrix [K=512 x N=512] (f32 Wv/Wg) into bf16 tiles in the
// exact WMMA B-operand swizzle (ISA 7.12.2, 16-bit 32x16 tile):
//   lane = (n&15) + 16*((kk>>3)&1), slot h = (kk&7) + 8*(kk>>4)
// Tile (kb,nb) = 1024 bytes, contiguous 32 bytes per lane.
// ---------------------------------------------------------------------------
__global__ void snail_pack_weights(const float* __restrict__ Wv,
                                   const float* __restrict__ Wg,
                                   unsigned short* __restrict__ bpack) {
    int id = blockIdx.x * blockDim.x + threadIdx.x;   // 0 .. K_*N_-1
    int k  = id >> 9;
    int n  = id & (N_ - 1);
    int tap = k >> 8;
    int c   = k & (C_ - 1);
    float src = (n < F_) ? Wv[tap * (C_ * F_) + c * F_ + n]
                         : Wg[tap * (C_ * F_) + c * F_ + (n - F_)];
    int kb = k >> 5, kk = k & 31, nb = n >> 4, nl = n & 15;
    int lane = nl + (((kk >> 3) & 1) << 4);
    int h    = (kk & 7) + ((kk >> 4) << 3);
    bpack[((kb * 32 + nb) << 9) + (lane << 4) + h] = f2bf(src);
}

union AF { v16bf bf; unsigned u[8]; };
union BF { v16bf bf; uint4 q[2]; };

// One K-step (32 channels): build bf16 A fragment from f32 row (masked,
// branch-free), batch-load 8 B fragments into distinct regs, 8 WMMAs.
__device__ __forceinline__ void kstep(const float* __restrict__ rp, int cbase,
                                      unsigned mk,
                                      const unsigned short* __restrict__ b0,
                                      int koff_us, v8f (&acc)[8]) {
    float4 f0 = *(const float4*)(rp + cbase);
    float4 f1 = *(const float4*)(rp + cbase + 4);
    float4 f2 = *(const float4*)(rp + cbase + 16);
    float4 f3 = *(const float4*)(rp + cbase + 20);
    AF a;
    a.bf[0]  = (__bf16)f0.x; a.bf[1]  = (__bf16)f0.y;
    a.bf[2]  = (__bf16)f0.z; a.bf[3]  = (__bf16)f0.w;
    a.bf[4]  = (__bf16)f1.x; a.bf[5]  = (__bf16)f1.y;
    a.bf[6]  = (__bf16)f1.z; a.bf[7]  = (__bf16)f1.w;
    a.bf[8]  = (__bf16)f2.x; a.bf[9]  = (__bf16)f2.y;
    a.bf[10] = (__bf16)f2.z; a.bf[11] = (__bf16)f2.w;
    a.bf[12] = (__bf16)f3.x; a.bf[13] = (__bf16)f3.y;
    a.bf[14] = (__bf16)f3.z; a.bf[15] = (__bf16)f3.w;
    #pragma unroll
    for (int q = 0; q < 8; ++q) a.u[q] &= mk;     // causal left-pad, no branch

    BF bfr[8];
    #pragma unroll
    for (int j = 0; j < 8; ++j) {
        // value tiles j=0..3 (nb = 4w+j), gate tiles j=4..7 (nb = 16+4w+j-4)
        const uint4* bp = (const uint4*)(b0 + koff_us + ((j & 3) << 9)
                                            + ((j >> 2) << 13));
        bfr[j].q[0] = bp[0];
        bfr[j].q[1] = bp[1];
    }
    #pragma unroll
    for (int j = 0; j < 8; ++j)
        acc[j] = __builtin_amdgcn_wmma_f32_16x16x32_bf16(
            false, a.bf, false, bfr[j].bf, (short)0, acc[j], false, false);
}

// 256 threads = 8 wave32. Workgroup tile: 16 rows x N=512; wave w owns f in
// [64w, 64w+64) for BOTH value and gate -> tanh*sigmoid fused in-register.
__global__ __launch_bounds__(256)
void snail_gemm(const float* __restrict__ x,
                const unsigned short* __restrict__ bpack,
                const float* __restrict__ bv,
                const float* __restrict__ bg,
                const int* __restrict__ dil,
                float* __restrict__ out) {
    const int d    = dil[0];
    const int lane = threadIdx.x & 31;
    const int w    = threadIdx.x >> 5;
    const long m0  = (long)blockIdx.x * 16;

    const int  arow = (int)m0 + (lane & 15);   // A row (both lane halves: same rows)
    const int  t    = arow & (T_ - 1);         // T is a power of two
    const int  ksel = lane >> 4;
    const bool havePrev = (t >= d);
    const float* xcur   = x + (long)arow * C_;
    const float* rpPrev = havePrev ? (xcur - (long)d * C_) : xcur;  // always valid
    const unsigned maskPrev = havePrev ? 0xFFFFFFFFu : 0u;

    const unsigned short* b0 =
        bpack + ((unsigned)(w * 4) << 9) + ((unsigned)lane << 4);

    v8f acc[8];
    #pragma unroll
    for (int j = 0; j < 8; ++j) acc[j] = (v8f){};

    // K halves fully unrolled: straight-line body lets the scheduler overlap
    // the next step's global loads with this step's WMMAs.
    #pragma unroll
    for (int kb = 0; kb < 8; ++kb)
        kstep(rpPrev, (kb << 5) + (ksel << 3), maskPrev, b0, kb * 16384, acc);
    #pragma unroll
    for (int kb = 0; kb < 8; ++kb)
        kstep(xcur, (kb << 5) + (ksel << 3), 0xFFFFFFFFu, b0,
              (kb + 8) * 16384, acc);

    // Epilogue: bias + tanh(v)*sigmoid(g); sigmoid(g) = 0.5 + 0.5*tanh(g/2).
    // C/D layout: VGPR r -> (M = r or r+8, N = lane&15).
    // Non-temporal stores: out is never re-read; don't evict x from L2.
    const int nl   = lane & 15;
    const int mloc = (lane >> 4) << 3;
    #pragma unroll
    for (int j = 0; j < 4; ++j) {
        const int f = w * 64 + j * 16 + nl;
        const float bvf = bv[f];
        const float bgf = bg[f];
        float* orow = out + (m0 + mloc) * OUTW + f;
        #pragma unroll
        for (int r = 0; r < 8; ++r) {
            float vv  = acc[j][r] + bvf;
            float gg  = acc[4 + j][r] + bgf;
            float act = fast_tanh(vv) * (0.5f + 0.5f * fast_tanh(0.5f * gg));
            __builtin_nontemporal_store(act, orow + (long)r * OUTW);
        }
    }

    // Identity half: out[row, 256:512] = x[row, :], coalesced float4 stream,
    // also non-temporal on the store side.
    #pragma unroll
    for (int i = 0; i < 4; ++i) {
        int idx = (int)threadIdx.x + (i << 8);   // 1024 float4 slots
        int rr  = idx >> 6;
        int cc  = (idx & 63) << 2;
        v4f v = *(const v4f*)(x + (m0 + rr) * C_ + cc);
        __builtin_nontemporal_store(v, (v4f*)(out + (m0 + rr) * OUTW + F_ + cc));
    }
}

extern "C" void kernel_launch(void* const* d_in, const int* in_sizes, int n_in,
                              void* d_out, int out_size, void* d_ws, size_t ws_size,
                              hipStream_t stream) {
    const float* x   = (const float*)d_in[0];
    const float* Wv  = (const float*)d_in[1];
    const float* bv  = (const float*)d_in[2];
    const float* Wg  = (const float*)d_in[3];
    const float* bg  = (const float*)d_in[4];
    const int*   dil = (const int*)d_in[5];
    float* out = (float*)d_out;
    unsigned short* bpack = (unsigned short*)d_ws;   // 512 KB used

    snail_pack_weights<<<(K_ * N_) / 256, 256, 0, stream>>>(Wv, Wg, bpack);
    snail_gemm<<<(B_ * T_) / 16, 256, 0, stream>>>(x, bpack, bv, bg, dil, out);
}
